// Vanilla_25443386262243
// MI455X (gfx1250) — matrix operand order repaired and checked by reading, not compiled, not run on hardware
//
#include <hip/hip_runtime.h>
#include <math.h>

// ---------------- types for CDNA5 WMMA ----------------
typedef __attribute__((ext_vector_type(8)))  __bf16 v8bf;
typedef __attribute__((ext_vector_type(16))) __bf16 v16bf;
typedef __attribute__((ext_vector_type(8)))  float  v8f;

// ---------------- problem constants ----------------
constexpr int kB  = 2;
constexpr int kS  = 2048;
constexpr int kD  = 1024;
constexpr int k3D = 3072;
constexpr int kH  = 16;
constexpr int kDH = 64;
constexpr int kRD = 32;
constexpr float kNegInf = -3.402823466e+38f;

// ---------------- bf16 helpers (RNE) ----------------
__device__ __forceinline__ unsigned short f2bf(float f) {
  unsigned u = __float_as_uint(f);
  unsigned r = u + 0x7FFFu + ((u >> 16) & 1u);
  return (unsigned short)(r >> 16);
}
__device__ __forceinline__ float bf2f(unsigned short h) {
  return __uint_as_float(((unsigned)h) << 16);
}

// ---------------- CDNA5 async global->LDS copy (16B per lane) ----------------
// GLOBAL_LOAD_ASYNC_TO_LDS_B128, GV mode: global vaddr copied to LDS address in vdst; ASYNCcnt.
// LDS byte offset = low 32 bits of the generic pointer (ISA aperture rule).
__device__ __forceinline__ void async_copy16(void* lds, const void* g) {
  unsigned d = (unsigned)(uintptr_t)lds;
  unsigned long long s = (unsigned long long)(uintptr_t)g;
  asm volatile("global_load_async_to_lds_b128 %0, %1, off"
               :: "v"(d), "v"(s) : "memory");
}
__device__ __forceinline__ void async_wait0() {
  asm volatile("s_wait_asynccnt 0x0" ::: "memory");
}

// ---------------- WMMA fragment loaders ----------------
// 16-bit A matrix 16x32 (per ISA): lanes 0-15 hold K=0..7 (v0..3) and
// K=16..23 (v4..7); lanes 16-31 hold K=8..15 and K=24..31.
__device__ __forceinline__ v16bf frag_a(const unsigned short* row, int kk, int half) {
  v8bf lo = *(const v8bf*)(row + kk + half * 8);
  v8bf hi = *(const v8bf*)(row + kk + 16 + half * 8);
  return __builtin_shufflevector(lo, hi, 0,1,2,3,4,5,6,7,8,9,10,11,12,13,14,15);
}
// 16-bit B matrix 32x16: lane n holds column n; lanes 0-15 K=0..7 and
// K=16..23, lanes 16-31 K=8..15 and K=24..31 (same k order as A). LDS stores
// B transposed ([n][k]) so a column is contiguous.
__device__ __forceinline__ v16bf frag_b(const unsigned short* col, int kk, int half) {
  v8bf lo = *(const v8bf*)(col + kk + half * 8);
  v8bf hi = *(const v8bf*)(col + kk + 16 + half * 8);
  return __builtin_shufflevector(lo, hi, 0,1,2,3,4,5,6,7,8,9,10,11,12,13,14,15);
}
__device__ __forceinline__ v8f wmma_bf16(v16bf a, v16bf b, v8f c) {
  return __builtin_amdgcn_wmma_f32_16x16x32_bf16(false, a, false, b, (short)0, c,
                                                 false, false);
}

// ---------------- kernel 1: f32 -> bf16 convert ----------------
__global__ void cvt_kernel(const float* __restrict__ src,
                           unsigned short* __restrict__ dst, int n) {
  int i = blockIdx.x * 256 + threadIdx.x;
  if (i < n) dst[i] = f2bf(src[i]);
}

// ---------------- kernel 2: LayerNorm -> bf16 ----------------
__global__ void ln_kernel(const float* __restrict__ X,
                          const float* __restrict__ gamma,
                          const float* __restrict__ beta,
                          unsigned short* __restrict__ XN) {
  const int row = blockIdx.x;
  const int t = threadIdx.x;
  const float* x = X + (size_t)row * kD;
  float vals[4];
  float s = 0.f, s2 = 0.f;
  for (int i = 0; i < 4; ++i) {
    float v = x[t + i * 256];
    vals[i] = v; s += v; s2 += v * v;
  }
  __shared__ float red[256], red2[256];
  red[t] = s; red2[t] = s2;
  __syncthreads();
  for (int off = 128; off > 0; off >>= 1) {
    if (t < off) { red[t] += red[t + off]; red2[t] += red2[t + off]; }
    __syncthreads();
  }
  float mean = red[0] * (1.0f / kD);
  float var  = red2[0] * (1.0f / kD) - mean * mean;
  float rstd = rsqrtf(var + 1e-5f);
  for (int i = 0; i < 4; ++i) {
    int d = t + i * 256;
    float xn = (vals[i] - mean) * rstd * gamma[d] + beta[d];
    XN[(size_t)row * kD + d] = f2bf(xn);
  }
}

// ---------------- kernel 3: QKV GEMM (WMMA, 64x128 tile, 4 wmma/wave/step) --
__global__ void gemm_qkv_kernel(const unsigned short* __restrict__ A,
                                const unsigned short* __restrict__ Bw,
                                const float* __restrict__ bias,
                                unsigned short* __restrict__ Qo,
                                unsigned short* __restrict__ Ko,
                                unsigned short* __restrict__ Vo) {
  __shared__ unsigned short As[64][32];    // [m][k]   4KB (filled by async DMA)
  __shared__ unsigned short Bs[128][32];   // [n][k]   8KB (transposed)
  const int tid = threadIdx.x;
  const int wave = tid >> 5, lane = tid & 31, half = lane >> 4, lm = lane & 15;
  const int wm = wave >> 2, wn = wave & 3;     // waves: 2 (M) x 4 (N)
  const int m0 = blockIdx.y * 64;
  const int n0 = blockIdx.x * 128;
  v8f acc00 = {}, acc01 = {}, acc10 = {}, acc11 = {};
  for (int k0 = 0; k0 < kD; k0 += 32) {
    // A tile: 64x32 bf16 via async global->LDS (one b128 per lane)
    {
      int r = tid >> 2, c = (tid & 3) * 8;
      async_copy16(&As[r][c], A + (size_t)(m0 + r) * kD + k0 + c);
    }
    // B tile: transpose-scatter into [n][k]
    {
      int i = tid * 16;
      int kk = i >> 7, nn = i & 127;
      const unsigned short* src = Bw + (size_t)(k0 + kk) * k3D + n0 + nn;
      unsigned short tmp[16];
      *(uint4*)&tmp[0] = *(const uint4*)(src);
      *(uint4*)&tmp[8] = *(const uint4*)(src + 8);
      if (k0 + 32 < kD) __builtin_prefetch(src + (size_t)32 * k3D, 0, 0);
      for (int u = 0; u < 16; ++u) Bs[nn + u][kk] = tmp[u];
    }
    async_wait0();
    __syncthreads();
    v16bf a0 = frag_a(&As[wm * 32 + lm][0], 0, half);
    v16bf a1 = frag_a(&As[wm * 32 + 16 + lm][0], 0, half);
    v16bf b0 = frag_b(&Bs[wn * 32 + lm][0], 0, half);
    v16bf b1 = frag_b(&Bs[wn * 32 + 16 + lm][0], 0, half);
    acc00 = wmma_bf16(a0, b0, acc00);
    acc01 = wmma_bf16(a0, b1, acc01);
    acc10 = wmma_bf16(a1, b0, acc10);
    acc11 = wmma_bf16(a1, b1, acc11);
    __syncthreads();
  }
  for (int mi = 0; mi < 2; ++mi) {
    for (int ni = 0; ni < 2; ++ni) {
      v8f acc = (mi == 0) ? (ni == 0 ? acc00 : acc01) : (ni == 0 ? acc10 : acc11);
      for (int r = 0; r < 8; ++r) {
        int m = m0 + wm * 32 + mi * 16 + half * 8 + r;
        int c = n0 + wn * 32 + ni * 16 + lm;
        float v = acc[r] + bias[c];
        int which = c >> 10;
        int d = c & 1023;
        int h = d >> 6, dh = d & 63;
        int bb = m >> 11, s = m & 2047;
        size_t idx = (((size_t)bb * kH + h) * kS + s) * kDH + dh;
        unsigned short* dst = (which == 0) ? Qo : ((which == 1) ? Ko : Vo);
        dst[idx] = f2bf(v);
      }
    }
  }
}

// ---------------- kernel 4: RoPE on Q,K ----------------
__global__ void rope_kernel(unsigned short* __restrict__ Q,
                            unsigned short* __restrict__ K) {
  int idx = blockIdx.x * 256 + threadIdx.x;
  if (idx >= kB * kH * kS * (kRD / 2)) return;
  int j  = idx & 15;
  int s  = (idx >> 4) & (kS - 1);
  int bh = idx >> 15;
  float inv = __powf(10000.0f, -(2.0f * (float)j) / (float)kRD);
  float ang = (float)s * inv;
  float cs = __cosf(ang), sn = __sinf(ang);
  size_t base = ((size_t)bh * kS + s) * kDH + 2 * j;
  {
    float x0 = bf2f(Q[base]), x1 = bf2f(Q[base + 1]);
    Q[base]     = f2bf(x0 * cs - x1 * sn);
    Q[base + 1] = f2bf(x1 * cs + x0 * sn);
  }
  {
    float x0 = bf2f(K[base]), x1 = bf2f(K[base + 1]);
    K[base]     = f2bf(x0 * cs - x1 * sn);
    K[base + 1] = f2bf(x1 * cs + x0 * sn);
  }
}

// ---------------- kernel 5: flash attention (WMMA) ----------------
// One block = one (b,h) head, 64 query rows. Online softmax over 32 key tiles.
__global__ void attn_kernel(const unsigned short* __restrict__ Q,
                            const unsigned short* __restrict__ K,
                            const unsigned short* __restrict__ V,
                            unsigned short* __restrict__ CTX) {
  __shared__ unsigned short Qs[64][64];   // [q][dh]   (async DMA)
  __shared__ unsigned short Ks[64][64];   // [key][dh] (async DMA; BT for QK^T)
  __shared__ unsigned short VTs[64][64];  // [dh][key] (BT for P.V)
  __shared__ float          Ss[64][64];   // scores
  __shared__ unsigned short Ps[64][64];   // softmax probs (bf16)
  __shared__ float rowM[64], rowL[64], rowA[64];

  const int tid = threadIdx.x;
  const int wave = tid >> 5, lane = tid & 31, half = lane >> 4, lm = lane & 15;
  const int wm = wave >> 1, wn = wave & 1;
  const int bh = blockIdx.y;
  const int q0 = blockIdx.x * 64;
  const size_t hbase = (size_t)bh * kS * kDH;

  {
    int r = tid >> 2, c = (tid & 3) * 16;
    const unsigned short* src = Q + hbase + (size_t)(q0 + r) * kDH + c;
    async_copy16(&Qs[r][c], src);
    async_copy16(&Qs[r][c + 8], src + 8);
  }
  if (tid < 64) { rowM[tid] = kNegInf; rowL[tid] = 0.0f; }
  v8f o0 = {}, o1 = {};
  async_wait0();
  __syncthreads();

  for (int kt = 0; kt < kS / 64; ++kt) {
    const size_t kbase = hbase + (size_t)kt * 64 * kDH;
    {
      int r = tid >> 2, c = (tid & 3) * 16;
      const unsigned short* ksrc = K + kbase + (size_t)r * kDH + c;
      async_copy16(&Ks[r][c], ksrc);
      async_copy16(&Ks[r][c + 8], ksrc + 8);
      unsigned short tmp[16];
      const unsigned short* vsrc = V + kbase + (size_t)r * kDH + c;
      *(uint4*)&tmp[0] = *(const uint4*)(vsrc);
      *(uint4*)&tmp[8] = *(const uint4*)(vsrc + 8);
      for (int i = 0; i < 16; ++i) VTs[c + i][r] = tmp[i];
    }
    async_wait0();
    __syncthreads();

    // scores: S = (Q K^T) * 1/sqrt(DH)
    v8f s0 = {}, s1 = {};
    for (int kk = 0; kk < 64; kk += 32) {
      v16bf aq = frag_a(&Qs[wm * 16 + lm][0], kk, half);
      v16bf b0 = frag_b(&Ks[wn * 16 + lm][0], kk, half);
      v16bf b1 = frag_b(&Ks[(wn + 2) * 16 + lm][0], kk, half);
      s0 = wmma_bf16(aq, b0, s0);
      s1 = wmma_bf16(aq, b1, s1);
    }
    for (int r = 0; r < 8; ++r) {
      int mr = wm * 16 + half * 8 + r;
      Ss[mr][wn * 16 + lm]       = s0[r] * 0.125f;
      Ss[mr][(wn + 2) * 16 + lm] = s1[r] * 0.125f;
    }
    __syncthreads();

    // online softmax, one thread per query row
    if (tid < 64) {
      float mo = rowM[tid];
      float mx = mo;
      for (int c = 0; c < 64; ++c) mx = fmaxf(mx, Ss[tid][c]);
      float al = __expf(mo - mx);
      float sum = 0.0f;
      for (int c = 0; c < 64; ++c) {
        float p = __expf(Ss[tid][c] - mx);
        Ps[tid][c] = f2bf(p);
        sum += p;
      }
      rowL[tid] = rowL[tid] * al + sum;
      rowM[tid] = mx;
      rowA[tid] = al;
    }
    __syncthreads();

    // rescale accumulators, then O += P @ V
    for (int r = 0; r < 8; ++r) {
      float al = rowA[wm * 16 + half * 8 + r];
      o0[r] *= al;
      o1[r] *= al;
    }
    for (int kk = 0; kk < 64; kk += 32) {
      v16bf ap = frag_a(&Ps[wm * 16 + lm][0], kk, half);
      v16bf b0 = frag_b(&VTs[wn * 16 + lm][0], kk, half);
      v16bf b1 = frag_b(&VTs[(wn + 2) * 16 + lm][0], kk, half);
      o0 = wmma_bf16(ap, b0, o0);
      o1 = wmma_bf16(ap, b1, o1);
    }
    __syncthreads();
  }

  const int b = bh / kH, h = bh % kH;
  for (int r = 0; r < 8; ++r) {
    int mr = wm * 16 + half * 8 + r;
    float linv = 1.0f / rowL[mr];
    int qg = q0 + mr;
    size_t o = ((size_t)b * kS + qg) * kD + (size_t)h * kDH;
    CTX[o + wn * 16 + lm]       = f2bf(o0[r] * linv);
    CTX[o + (wn + 2) * 16 + lm] = f2bf(o1[r] * linv);
  }
}

// ---------------- kernel 6: output GEMM + bias + residual ----------------
__global__ void gemm_out_kernel(const unsigned short* __restrict__ A,
                                const unsigned short* __restrict__ Bw,
                                const float* __restrict__ bias,
                                const float* __restrict__ X,
                                float* __restrict__ Out) {
  __shared__ unsigned short As[64][32];
  __shared__ unsigned short Bs[128][32];
  const int tid = threadIdx.x;
  const int wave = tid >> 5, lane = tid & 31, half = lane >> 4, lm = lane & 15;
  const int wm = wave >> 2, wn = wave & 3;
  const int m0 = blockIdx.y * 64;
  const int n0 = blockIdx.x * 128;
  v8f acc00 = {}, acc01 = {}, acc10 = {}, acc11 = {};
  for (int k0 = 0; k0 < kD; k0 += 32) {
    {
      int r = tid >> 2, c = (tid & 3) * 8;
      async_copy16(&As[r][c], A + (size_t)(m0 + r) * kD + k0 + c);
    }
    {
      int i = tid * 16;
      int kk = i >> 7, nn = i & 127;
      const unsigned short* src = Bw + (size_t)(k0 + kk) * kD + n0 + nn;
      unsigned short tmp[16];
      *(uint4*)&tmp[0] = *(const uint4*)(src);
      *(uint4*)&tmp[8] = *(const uint4*)(src + 8);
      if (k0 + 32 < kD) __builtin_prefetch(src + (size_t)32 * kD, 0, 0);
      for (int u = 0; u < 16; ++u) Bs[nn + u][kk] = tmp[u];
    }
    async_wait0();
    __syncthreads();
    v16bf a0 = frag_a(&As[wm * 32 + lm][0], 0, half);
    v16bf a1 = frag_a(&As[wm * 32 + 16 + lm][0], 0, half);
    v16bf b0 = frag_b(&Bs[wn * 32 + lm][0], 0, half);
    v16bf b1 = frag_b(&Bs[wn * 32 + 16 + lm][0], 0, half);
    acc00 = wmma_bf16(a0, b0, acc00);
    acc01 = wmma_bf16(a0, b1, acc01);
    acc10 = wmma_bf16(a1, b0, acc10);
    acc11 = wmma_bf16(a1, b1, acc11);
    __syncthreads();
  }
  for (int mi = 0; mi < 2; ++mi) {
    for (int ni = 0; ni < 2; ++ni) {
      v8f acc = (mi == 0) ? (ni == 0 ? acc00 : acc01) : (ni == 0 ? acc10 : acc11);
      for (int r = 0; r < 8; ++r) {
        int m = m0 + wm * 32 + mi * 16 + half * 8 + r;
        int c = n0 + wn * 32 + ni * 16 + lm;
        size_t idx = (size_t)m * kD + c;
        Out[idx] = acc[r] + bias[c] + X[idx];
      }
    }
  }
}

// ---------------- host launcher ----------------
extern "C" void kernel_launch(void* const* d_in, const int* in_sizes, int n_in,
                              void* d_out, int out_size, void* d_ws, size_t ws_size,
                              hipStream_t stream) {
  (void)in_sizes; (void)n_in; (void)out_size; (void)ws_size;
  const float* X     = (const float*)d_in[0];
  const float* W_in  = (const float*)d_in[1];
  const float* b_in  = (const float*)d_in[2];
  const float* W_out = (const float*)d_in[3];
  const float* b_out = (const float*)d_in[4];
  const float* gamma = (const float*)d_in[5];
  const float* beta  = (const float*)d_in[6];
  float* out = (float*)d_out;

  char* ws = (char*)d_ws;
  unsigned short* XN   = (unsigned short*)(ws + 0);          //  8 MB
  unsigned short* WIN  = (unsigned short*)(ws + 8388608);    //  6 MB
  unsigned short* WOUT = (unsigned short*)(ws + 14680064);   //  2 MB
  unsigned short* Qb   = (unsigned short*)(ws + 16777216);   //  8 MB
  unsigned short* Kb   = (unsigned short*)(ws + 25165824);   //  8 MB
  unsigned short* Vb   = (unsigned short*)(ws + 33554432);   //  8 MB
  unsigned short* CTX  = (unsigned short*)(ws + 41943040);   //  8 MB

  // 1. weights -> bf16
  cvt_kernel<<<(kD * k3D + 255) / 256, 256, 0, stream>>>(W_in, WIN, kD * k3D);
  cvt_kernel<<<(kD * kD + 255) / 256, 256, 0, stream>>>(W_out, WOUT, kD * kD);
  // 2. LayerNorm
  ln_kernel<<<kB * kS, 256, 0, stream>>>(X, gamma, beta, XN);
  // 3. QKV projection
  gemm_qkv_kernel<<<dim3(k3D / 128, (kB * kS) / 64), 256, 0, stream>>>(
      XN, WIN, b_in, Qb, Kb, Vb);
  // 4. RoPE
  rope_kernel<<<(kB * kH * kS * (kRD / 2) + 255) / 256, 256, 0, stream>>>(Qb, Kb);
  // 5. Flash attention
  attn_kernel<<<dim3(kS / 64, kB * kH), 256, 0, stream>>>(Qb, Kb, Vb, CTX);
  // 6. Output projection + residual
  gemm_out_kernel<<<dim3(kD / 128, (kB * kS) / 64), 256, 0, stream>>>(
      CTX, WOUT, b_out, X, out);
}
